// MoELayer_23433341567138
// MI455X (gfx1250) — compile-verified, matrix operand order checked by reading, not run on hardware
//
#include <hip/hip_runtime.h>
#include <math.h>

// ---------------- CDNA5 WMMA types ----------------
typedef __attribute__((ext_vector_type(16))) __bf16 v16bf;
typedef __attribute__((ext_vector_type(8)))  float  v8f;
typedef unsigned int u32x4 __attribute__((ext_vector_type(4)));
typedef int          i32x4 __attribute__((ext_vector_type(4)));
typedef int          i32x8 __attribute__((ext_vector_type(8)));

#if defined(__has_builtin)
#if __has_builtin(__builtin_amdgcn_tensor_load_to_lds) && __has_builtin(__builtin_amdgcn_s_wait_tensorcnt)
#define HAVE_TDM 1
#endif
#endif

#define DIMC  128
#define HEADS 8
#define NEXP  7
#define TOPK  4
#define BATCH 8
#define HH    64
#define WWW   64
#define HWN   4096            // 64*64
#define CHW   (DIMC*HWN)      // 524288
#define MAXK  256             // largest GEMM K (contrast diff conv)

__device__ __forceinline__ float geluf(float v){ return 0.5f*v*(1.0f+erff(v*0.70710678118f)); }
__device__ __forceinline__ float sigm(float v){ return 1.0f/(1.0f+__expf(-v)); }

// f32 -> bf16 (round-to-nearest-even), stored as ushort
__device__ __forceinline__ unsigned short f2b(float f){
  union { float f; unsigned u; } u; u.f = f;
  unsigned r = u.u + 0x7FFFu + ((u.u >> 16) & 1u);
  return (unsigned short)(r >> 16);
}

// K index map for 16-bit A/B fragments of V_WMMA_*_16X16X32 (wave32):
// VGPR-pair v (0..7), sub (0/1), half = lane>>4
__device__ __forceinline__ int kmap(int v, int sub, int half){
  return ((v & 4) << 2) + half * 8 + (v & 3) * 2 + sub;
}

// ---- Tensor Data Mover: 2D tile (rows x elems_per_row of bf16) global -> LDS ----
// D# layout per CDNA5 ISA ch.8: group0 = {flags, lds_addr, global_addr, type=2},
// group1 = {data_size=2B, tensor dims, tile dims, dim0 stride}. groups 2/3 zero (2D).
__device__ __forceinline__ void tdm_load_2d(unsigned lds_off, const void* gaddr,
                                            int elems_per_row, int rows,
                                            int row_stride_elems, int tensor_rows){
#ifdef HAVE_TDM
  unsigned long long ga = (unsigned long long)gaddr;
  u32x4 g0;
  g0.x = 1u;                                                   // count=1, user mode
  g0.y = lds_off;                                              // LDS byte address
  g0.z = (unsigned)ga;                                         // global_addr[31:0]
  g0.w = ((unsigned)(ga >> 32) & 0x01FFFFFFu) | 0x80000000u;   // addr[56:32] | type=2
  unsigned td0 = (unsigned)row_stride_elems;                   // tensor_dim0
  unsigned td1 = (unsigned)tensor_rows;                        // tensor_dim1
  i32x8 g1;
  g1[0] = 1 << 16;                                             // data_size = 2 bytes
  g1[1] = (int)((td0 & 0xFFFFu) << 16);                        // tensor_dim0 lo
  g1[2] = (int)((td0 >> 16) | ((td1 & 0xFFFFu) << 16));        // dim0 hi | dim1 lo
  g1[3] = (int)((td1 >> 16) | ((unsigned)elems_per_row << 16));// dim1 hi | tile_dim0
  g1[4] = (int)rows;                                           // tile_dim1 (tile_dim2=0)
  g1[5] = (int)row_stride_elems;                               // tensor_dim0_stride lo
  g1[6] = 0; g1[7] = 0;
  i32x4 z4 = (i32x4)0;
#if __clang_major__ >= 23
  i32x8 z8 = (i32x8)0;
  __builtin_amdgcn_tensor_load_to_lds(g0, g1, z4, z4, z8, 0);
#else
  __builtin_amdgcn_tensor_load_to_lds(g0, g1, z4, z4, 0);
#endif
#else
  (void)lds_off; (void)gaddr; (void)elems_per_row; (void)rows;
  (void)row_stride_elems; (void)tensor_rows;
#endif
}

// ---------------- fast 1x1-conv GEMM, bf16 WMMA, f32 accumulate ----------------
// Requires N%32==0, Cout%16==0, K%32==0, K<=MAXK. One wave computes a 16x32 tile:
// full A(16xK) and B(Kx32) panels staged in LDS (TDM when available), then a
// barrier-free ds_load + 2x v_wmma inner loop.
__global__ __launch_bounds__(32)
void k_gemm(const unsigned short* __restrict__ Wm,   // [Cout x K] bf16
            const unsigned short* __restrict__ X,    // [B][K x N] bf16
            const float* __restrict__ bias,          // [Cout] or null
            float* __restrict__ Y,                   // [B][yctot x N]
            unsigned short* __restrict__ Yb,         // optional bf16 mirror
            int K, int Cout, int N, int xbs, int coff, int yctot, int act)
{
  __shared__ unsigned short Tile[16*MAXK + MAXK*32]; // A panel @0, B panel @16*MAXK
  unsigned short* At = Tile;
  unsigned short* Bt = Tile + 16*MAXK;
  int lane = threadIdx.x;
  int half = lane >> 4, idx = lane & 15;
  int b  = blockIdx.z;
  int Mb = blockIdx.y * 16;
  int Nb = blockIdx.x * 32;
  const unsigned short* Xb = X + (size_t)b * xbs;
#ifdef HAVE_TDM
  // Real LDS byte offsets of the panels: low 32 bits of the flat LDS address
  // (ISA 10.2: LDS aperture maps addr[31:0] -> LDS offset). The ptrtoint also
  // escapes the shared array so the optimizer must honor TDM's writes to it.
  unsigned aoff = (unsigned)(unsigned long long)(void*)At;
  unsigned boff = (unsigned)(unsigned long long)(void*)Bt;
  if (lane == 0) {
    tdm_load_2d(aoff, Wm + (size_t)Mb * K, K, 16, K, Cout - Mb);        // A panel
    tdm_load_2d(boff, Xb + Nb, 32, K, N, K);                            // B panel
  }
  __builtin_amdgcn_s_wait_tensorcnt(0);
  // Escape barrier: the asm "may write" through At/Bt, so the ds_load fragment
  // gathers below cannot be folded to undef and cannot be hoisted above the wait.
  asm volatile("" :: "v"(At), "v"(Bt) : "memory");
#else
  { // A panel: contiguous 16*K bf16 starting at row Mb
    const unsigned int* src = (const unsigned int*)(Wm + (size_t)Mb * K);
    unsigned int* dst = (unsigned int*)At;
    int tot = 16 * K / 2;
    for (int i2 = lane; i2 < tot; i2 += 32) dst[i2] = src[i2];
  }
  { // B panel: K rows x 32 cols
    unsigned int* dst = (unsigned int*)Bt;
    int tot = K * 16;
    for (int i2 = lane; i2 < tot; i2 += 32) {
      int row = i2 >> 4, col = i2 & 15;
      dst[i2] = ((const unsigned int*)(Xb + (size_t)row * N + Nb))[col];
    }
  }
#endif
  __syncthreads();
  v8f acc0 = {}, acc1 = {};
  for (int k0 = 0; k0 < K; k0 += 32) {
    union { v16bf v; unsigned int u[8]; } af;
    #pragma unroll
    for (int vv = 0; vv < 8; vv++)
      af.u[vv] = *(const unsigned int*)(At + (size_t)idx * K + k0 + kmap(vv, 0, half));
    union { v16bf v; unsigned short s[16]; } b0, b1;
    #pragma unroll
    for (int e = 0; e < 16; e++) {
      int kk = k0 + kmap(e >> 1, e & 1, half);
      b0.s[e] = Bt[kk * 32 + idx];
      b1.s[e] = Bt[kk * 32 + 16 + idx];
    }
    acc0 = __builtin_amdgcn_wmma_f32_16x16x32_bf16(false, af.v, false, b0.v,
                                                   (short)0, acc0, false, false);
    acc1 = __builtin_amdgcn_wmma_f32_16x16x32_bf16(false, af.v, false, b1.v,
                                                   (short)0, acc1, false, false);
  }
  #pragma unroll
  for (int r = 0; r < 8; r++) {
    int m = r + half * 8;
    int cg = Mb + m;
    float base = bias ? bias[cg] : 0.0f;
    float v0 = acc0[r] + base, v1 = acc1[r] + base;
    if (act == 1) { v0 = geluf(v0); v1 = geluf(v1); }
    else if (act == 2) { v0 = sigm(v0); v1 = sigm(v1); }
    size_t o = ((size_t)b * yctot + coff + cg) * (size_t)N + Nb + idx;
    Y[o] = v0; Y[o + 16] = v1;
    if (Yb) { Yb[o] = f2b(v0); Yb[o + 16] = f2b(v1); }
  }
}

// tiny-N GEMM (semantic expert 4x4/2x2/1x1 pools) — plain f32, no guards needed
__global__ void k_gemm_small(const float* __restrict__ Wf, const float* __restrict__ X,
                             const float* __restrict__ bias, float* __restrict__ Y,
                             int K, int Cout, int N, int act){
  int i = blockIdx.x * blockDim.x + threadIdx.x;
  int tot = BATCH * Cout * N;
  if (i >= tot) return;
  int n = i % N; int m = (i / N) % Cout; int b = i / (Cout * N);
  const float* xb = X + (size_t)b * K * N;
  float s = bias ? bias[m] : 0.0f;
  for (int k = 0; k < K; k++) s += Wf[(size_t)m * K + k] * xb[(size_t)k * N + n];
  if (act == 1) s = geluf(s); else if (act == 2) s = sigm(s);
  Y[((size_t)b * Cout + m) * N + n] = s;
}

// ---------------- attention expert core (per block: one (b,head)) ----------------
__global__ __launch_bounds__(32)
void k_attn(const unsigned short* __restrict__ qkvb,  // [B][384*HWN] bf16
            float* __restrict__ O, unsigned short* __restrict__ Ob) // [B][128*HWN]
{
  __shared__ float sc[16*16];
  __shared__ unsigned short at[16*16];
  int lane = threadIdx.x;
  int half = lane >> 4, idx = lane & 15;
  int h = blockIdx.x, b = blockIdx.y;
  const unsigned short* Q  = qkvb + (size_t)b * (3*CHW) + (size_t)(h * 16) * HWN;
  const unsigned short* Kp = Q + (size_t)DIMC * HWN;
  const unsigned short* V  = Q + (size_t)(2 * DIMC) * HWN;
  v8f acc = {};
  for (int n0 = 0; n0 < HWN; n0 += 32) {           // K=4096 reduction via WMMA
    union { v16bf v; unsigned int u[8]; } af, bq;
    #pragma unroll
    for (int vv = 0; vv < 8; vv++) {
      int kk = kmap(vv, 0, half);
      af.u[vv] = *(const unsigned int*)(Q  + (size_t)idx * HWN + n0 + kk);
      bq.u[vv] = *(const unsigned int*)(Kp + (size_t)idx * HWN + n0 + kk);
    }
    if (n0 + 32 < HWN) {
      __builtin_prefetch(Q  + (size_t)idx * HWN + n0 + 32, 0, 3);
      __builtin_prefetch(Kp + (size_t)idx * HWN + n0 + 32, 0, 3);
    }
    acc = __builtin_amdgcn_wmma_f32_16x16x32_bf16(false, af.v, false, bq.v,
                                                  (short)0, acc, false, false);
  }
  #pragma unroll
  for (int r = 0; r < 8; r++) { int m = r + half * 8; sc[m*16 + idx] = acc[r] * (1.0f/64.0f); }
  __syncthreads();
  if (lane < 16) {                                  // row softmax
    float mx = -1e30f;
    for (int j = 0; j < 16; j++) mx = fmaxf(mx, sc[lane*16 + j]);
    float s = 0.0f;
    for (int j = 0; j < 16; j++) { float e = __expf(sc[lane*16 + j] - mx); sc[lane*16 + j] = e; s += e; }
    float inv = 1.0f / s;
    for (int j = 0; j < 16; j++) at[lane*16 + j] = f2b(sc[lane*16 + j] * inv);
  }
  __syncthreads();
  union { v16bf v; unsigned short s[16]; } afa;     // attn as A (K=16 padded to 32)
  #pragma unroll
  for (int e = 0; e < 16; e++) {
    int kk = kmap(e >> 1, e & 1, half);
    afa.s[e] = (kk < 16) ? at[idx * 16 + kk] : (unsigned short)0;
  }
  for (int n0 = 0; n0 < HWN; n0 += 16) {
    union { v16bf v; unsigned short s[16]; } bv;
    #pragma unroll
    for (int e = 0; e < 16; e++) {
      int kk = kmap(e >> 1, e & 1, half);
      bv.s[e] = (kk < 16) ? V[(size_t)kk * HWN + n0 + idx] : (unsigned short)0;
    }
    v8f o = {};
    o = __builtin_amdgcn_wmma_f32_16x16x32_bf16(false, afa.v, false, bv.v,
                                                (short)0, o, false, false);
    #pragma unroll
    for (int r = 0; r < 8; r++) {
      int m = r + half * 8;
      size_t off = (size_t)b * CHW + (size_t)(h * 16 + m) * HWN + n0 + idx;
      O[off] = o[r]; Ob[off] = f2b(o[r]);
    }
  }
}

// ---------------- elementwise / LN / pooling kernels ----------------
__global__ void k_pack(const float* __restrict__ X, unsigned short* __restrict__ Y, size_t n){
  size_t i = (size_t)blockIdx.x * blockDim.x + threadIdx.x;
  if (i < n) Y[i] = f2b(X[i]);
}

__global__ void k_pack_slice(const float* __restrict__ X, unsigned short* __restrict__ Y,
                             int C, int N, int coff, int Ctot){
  size_t i = (size_t)blockIdx.x * blockDim.x + threadIdx.x;
  size_t tot = (size_t)BATCH * C * N;
  if (i >= tot) return;
  int n = (int)(i % N); int c = (int)((i / N) % C); int b = (int)(i / ((size_t)C * N));
  Y[((size_t)b * Ctot + coff + c) * N + n] = f2b(X[i]);
}

__global__ void k_add(const float* __restrict__ A, const float* __restrict__ B2,
                      float* __restrict__ Y, size_t n){
  size_t i = (size_t)blockIdx.x * blockDim.x + threadIdx.x;
  if (i < n) Y[i] = A[i] + B2[i];
}

// layer-norm across channels at each pixel; optional gelu; optional bf16 mirror;
// output written at channel offset ycoff inside a [B][yctot x N] tensor.
__global__ void k_ln2d(const float* __restrict__ X, const float* __restrict__ g,
                       const float* __restrict__ bb, float* __restrict__ Y,
                       unsigned short* __restrict__ Yb,
                       int C, int N, int act, int ycoff, int yctot){
  int p = blockIdx.x * blockDim.x + threadIdx.x;
  int total = BATCH * N;
  if (p >= total) return;
  int b = p / N, n = p % N;
  const float* xb = X + (size_t)b * C * N + n;
  float s = 0.f, s2 = 0.f;
  for (int c = 0; c < C; c++) { float v = xb[(size_t)c * N]; s += v; s2 += v * v; }
  float m = s / C;
  float inv = rsqrtf(s2 / C - m * m + 1e-6f);
  float* yo = Y + ((size_t)b * yctot + ycoff) * N + n;
  unsigned short* yb = Yb ? Yb + ((size_t)b * yctot + ycoff) * N + n : (unsigned short*)0;
  for (int c = 0; c < C; c++) {
    float v = (xb[(size_t)c * N] - m) * inv * g[c] + bb[c];
    if (act == 1) v = geluf(v);
    yo[(size_t)c * N] = v;
    if (yb) yb[(size_t)c * N] = f2b(v);
  }
}

__global__ void k_gap(const float* __restrict__ X, float* __restrict__ G, int C, int N){
  int i = blockIdx.x * blockDim.x + threadIdx.x;
  if (i >= BATCH * C) return;
  const float* xb = X + (size_t)i * N;
  float s = 0.f; for (int n = 0; n < N; n++) s += xb[n];
  G[i] = s / N;
}

// SE gate: gate[b][c] = sigmoid(W2·gelu(W1·g + b1) + b2)
__global__ void k_se(const float* __restrict__ G, const float* __restrict__ w1,
                     const float* __restrict__ b1, const float* __restrict__ w2,
                     const float* __restrict__ b2, float* __restrict__ gate, int C, int C1){
  __shared__ float h[32];
  int b = blockIdx.x, t = threadIdx.x;
  const float* g = G + (size_t)b * C;
  if (t < C1) { float s = b1[t]; for (int c = 0; c < C; c++) s += w1[t*C + c] * g[c]; h[t] = geluf(s); }
  __syncthreads();
  if (t < C) { float s = b2[t]; for (int c = 0; c < C1; c++) s += w2[t*C1 + c] * h[c]; gate[(size_t)b*C + t] = sigm(s); }
}

__global__ void k_scalech(const float* __restrict__ X, const float* __restrict__ gate,
                          float* __restrict__ Y, unsigned short* __restrict__ Yb, int C, int N){
  size_t i = (size_t)blockIdx.x * blockDim.x + threadIdx.x;
  size_t tot = (size_t)BATCH * C * N;
  if (i >= tot) return;
  int c = (int)((i / N) % C); int b = (int)(i / ((size_t)C * N));
  float v = X[i] * gate[(size_t)b * C + c];
  Y[i] = v; if (Yb) Yb[i] = f2b(v);
}

// learned depthwise conv (ks=3 pad1 or ks=5 pad2), zero padding
__global__ void k_dwconv(const float* __restrict__ X, const float* __restrict__ Wd,
                         const float* __restrict__ bias, float* __restrict__ Y, int ks, int pad){
  size_t i = (size_t)blockIdx.x * blockDim.x + threadIdx.x;
  if (i >= (size_t)BATCH * CHW) return;
  int n = (int)(i % HWN); int c = (int)((i / HWN) % DIMC); int b = (int)(i / CHW);
  int yy = n / WWW, xx = n % WWW;
  const float* xb = X + (size_t)b * CHW + (size_t)c * HWN;
  float s = bias ? bias[c] : 0.0f;
  for (int ky = 0; ky < ks; ky++) for (int kx = 0; kx < ks; kx++) {
    int sy = yy + ky - pad, sx = xx + kx - pad;
    if (sy >= 0 && sy < HH && sx >= 0 && sx < WWW)
      s += Wd[c*ks*ks + ky*ks + kx] * xb[sy*WWW + sx];
  }
  Y[i] = s;
}

// fixed sobel/laplace features -> bf16 directly (conv inputs)
__global__ void k_edgefeat(const float* __restrict__ X, unsigned short* __restrict__ SOB,
                           unsigned short* __restrict__ LAP, unsigned short* __restrict__ GRD){
  size_t i = (size_t)blockIdx.x * blockDim.x + threadIdx.x;
  if (i >= (size_t)BATCH * CHW) return;
  int n = (int)(i % HWN); int c = (int)((i / HWN) % DIMC); int b = (int)(i / CHW);
  int yy = n / WWW, xx = n % WWW;
  const float* xb = X + (size_t)b * CHW + (size_t)c * HWN;
  const float KX[9] = {-1,0,1,-2,0,2,-1,0,1};
  const float KY[9] = {-1,-2,-1,0,0,0,1,2,1};
  const float KL[9] = {0,1,0,1,-4,1,0,1,0};
  float sx = 0.f, sy = 0.f, lp = 0.f;
  for (int ky = 0; ky < 3; ky++) for (int kx = 0; kx < 3; kx++) {
    int py = yy + ky - 1, px = xx + kx - 1;
    if (py < 0 || py >= HH || px < 0 || px >= WWW) continue;
    float v = xb[py*WWW + px];
    sx += KX[ky*3+kx]*v; sy += KY[ky*3+kx]*v; lp += KL[ky*3+kx]*v;
  }
  float sob = sqrtf(sx*sx + sy*sy + 1e-8f);
  float lap = fabsf(lp);
  float grd = sqrtf(sob*sob + lap*lap + 1e-8f);
  SOB[i] = f2b(sob); LAP[i] = f2b(lap); GRD[i] = f2b(grd);
}

// low/mid/high frequency split -> bf16 (conv inputs)
__global__ void k_freqpre(const float* __restrict__ X, unsigned short* __restrict__ LOW,
                          unsigned short* __restrict__ MID, unsigned short* __restrict__ HIGH){
  size_t i = (size_t)blockIdx.x * blockDim.x + threadIdx.x;
  if (i >= (size_t)BATCH * CHW) return;
  int n = (int)(i % HWN); int c = (int)((i / HWN) % DIMC); int b = (int)(i / CHW);
  int yy = n / WWW, xx = n % WWW;
  const float* xb = X + (size_t)b * CHW + (size_t)c * HWN;
  float s3 = 0.f, s5 = 0.f;
  for (int ky = -2; ky <= 2; ky++) for (int kx = -2; kx <= 2; kx++) {
    int py = yy + ky, px = xx + kx;
    if (py < 0 || py >= HH || px < 0 || px >= WWW) continue;
    float v = xb[py*WWW + px];
    s5 += v;
    if (ky >= -1 && ky <= 1 && kx >= -1 && kx <= 1) s3 += v;
  }
  float low = s3 / 9.0f, avg5 = s5 / 25.0f, xv = xb[yy*WWW + xx];
  LOW[i] = f2b(low); MID[i] = f2b(low - avg5); HIGH[i] = f2b(xv - low);
}

__global__ void k_apool(const float* __restrict__ X, float* __restrict__ Y, int kk){
  int i = blockIdx.x * blockDim.x + threadIdx.x;
  int tot = BATCH * DIMC * kk * kk;
  if (i >= tot) return;
  int j = i % (kk*kk); int c = (i / (kk*kk)) % DIMC; int b = i / (DIMC*kk*kk);
  int bi = j / kk, bj = j % kk;
  int bh = HH / kk, bw = WWW / kk;
  const float* xb = X + (size_t)b * CHW + (size_t)c * HWN;
  float s = 0.f;
  for (int y = bi*bh; y < bi*bh + bh; y++)
    for (int x = bj*bw; x < bj*bw + bw; x++) s += xb[y*WWW + x];
  Y[i] = s / (bh * bw);
}

// bilinear resize (half-pixel) from [B][C2 x kk x kk] into feats channel slice
__global__ void k_resize(const float* __restrict__ X, int kk, int C2,
                         float* __restrict__ Y, unsigned short* __restrict__ Yb,
                         int coff, int Ctot){
  size_t i = (size_t)blockIdx.x * blockDim.x + threadIdx.x;
  size_t tot = (size_t)BATCH * C2 * HWN;
  if (i >= tot) return;
  int n = (int)(i % HWN); int c = (int)((i / HWN) % C2); int b = (int)(i / ((size_t)C2 * HWN));
  int yy = n / WWW, xx = n % WWW;
  float scale = (float)kk / 64.0f;
  float fy = (yy + 0.5f) * scale - 0.5f, fx = (xx + 0.5f) * scale - 0.5f;
  int y0 = (int)floorf(fy), x0 = (int)floorf(fx);
  float wy = fy - y0, wx = fx - x0;
  int y1 = y0 + 1, x1 = x0 + 1;
  y0 = min(max(y0,0),kk-1); y1 = min(max(y1,0),kk-1);
  x0 = min(max(x0,0),kk-1); x1 = min(max(x1,0),kk-1);
  const float* xb = X + ((size_t)b * C2 + c) * kk * kk;
  float v = (1-wy)*((1-wx)*xb[y0*kk+x0] + wx*xb[y0*kk+x1])
          +    wy *((1-wx)*xb[y1*kk+x0] + wx*xb[y1*kk+x1]);
  size_t off = ((size_t)b * Ctot + coff + c) * HWN + n;
  Y[off] = v; if (Yb) Yb[off] = f2b(v);
}

__global__ void k_absdiff(const float* __restrict__ A, const float* __restrict__ B2,
                          float* __restrict__ Y, size_t n){
  size_t i = (size_t)blockIdx.x * blockDim.x + threadIdx.x;
  if (i < n) Y[i] = fabsf(A[i] - B2[i]);
}

__global__ void k_diveps(const float* __restrict__ A, const float* __restrict__ S,
                         float* __restrict__ Y, size_t n){
  size_t i = (size_t)blockIdx.x * blockDim.x + threadIdx.x;
  if (i < n) Y[i] = A[i] / (S[i] + 1e-6f);
}

// dense 3x3 conv (contrast vis1/vis2), direct form
__global__ void k_conv3x3(const float* __restrict__ X, const float* __restrict__ Wc,
                          const float* __restrict__ bias, float* __restrict__ Y,
                          int Cin, int Cout){
  size_t i = (size_t)blockIdx.x * blockDim.x + threadIdx.x;
  size_t tot = (size_t)BATCH * Cout * HWN;
  if (i >= tot) return;
  int n = (int)(i % HWN); int co = (int)((i / HWN) % Cout); int b = (int)(i / ((size_t)Cout * HWN));
  int yy = n / HWN % 1 + n / WWW, xx = n % WWW;
  yy = n / WWW;
  float s = bias ? bias[co] : 0.0f;
  const float* xb = X + (size_t)b * Cin * HWN;
  for (int ci = 0; ci < Cin; ci++) {
    const float* w = Wc + ((size_t)co * Cin + ci) * 9;
    const float* xc = xb + (size_t)ci * HWN;
    for (int ky = 0; ky < 3; ky++) for (int kx = 0; kx < 3; kx++) {
      int sy = yy + ky - 1, sx = xx + kx - 1;
      if (sy >= 0 && sy < HH && sx >= 0 && sx < WWW)
        s += w[ky*3 + kx] * xc[sy*WWW + sx];
    }
  }
  Y[i] = s;
}

// router: gap(x) -> gelu(W1)+ -> logits -> top4 softmax -> dense[B][7]
__global__ void k_router(const float* __restrict__ G, const float* __restrict__ w1,
                         const float* __restrict__ b1, const float* __restrict__ w2,
                         const float* __restrict__ b2, float* __restrict__ dense){
  int b = threadIdx.x;
  if (b >= BATCH) return;
  const float* g = G + (size_t)b * DIMC;
  float hid[32];
  for (int t = 0; t < 32; t++) {
    float s = b1[t];
    for (int c = 0; c < DIMC; c++) s += w1[t*DIMC + c] * g[c];
    hid[t] = geluf(s);
  }
  float lg[NEXP];
  for (int e = 0; e < NEXP; e++) {
    float s = b2[e];
    for (int c = 0; c < 32; c++) s += w2[e*32 + c] * hid[c];
    lg[e] = s;
  }
  int sel[TOPK]; float val[TOPK]; bool used[NEXP] = {false,false,false,false,false,false,false};
  for (int t = 0; t < TOPK; t++) {
    int bi = -1; float bv = -1e30f;
    for (int e = 0; e < NEXP; e++) if (!used[e] && lg[e] > bv) { bv = lg[e]; bi = e; }
    used[bi] = true; sel[t] = bi; val[t] = bv;
  }
  float mx = val[0], s = 0.f;
  for (int t = 0; t < TOPK; t++) { val[t] = __expf(val[t] - mx); s += val[t]; }
  for (int e = 0; e < NEXP; e++) dense[(size_t)b*NEXP + e] = 0.f;
  for (int t = 0; t < TOPK; t++) dense[(size_t)b*NEXP + sel[t]] = val[t] / s;
}

__global__ void k_combine(const float* __restrict__ E, const float* __restrict__ dense,
                          float* __restrict__ O){
  size_t i = (size_t)blockIdx.x * blockDim.x + threadIdx.x;
  if (i >= (size_t)BATCH * CHW) return;
  int b = (int)(i / CHW);
  float s = 0.f;
  for (int e = 0; e < NEXP; e++)
    s += dense[(size_t)b*NEXP + e] * E[(size_t)e * BATCH * CHW + i];
  O[i] = s;
}

// ---------------- host orchestration ----------------
static inline int div_up(int a, int b){ return (a + b - 1) / b; }

extern "C" void kernel_launch(void* const* d_in, const int* in_sizes, int n_in,
                              void* d_out, int out_size, void* d_ws, size_t ws_size,
                              hipStream_t stream) {
  (void)in_sizes; (void)n_in; (void)out_size; (void)ws_size;
  int gi = 0;
  auto P = [&]() -> const float* { return (const float*)d_in[gi++]; };

  const float* x = P();
  // router
  const float *r_w1=P(),*r_b1=P(),*r_w2=P(),*r_b2=P();
  // attn
  const float *a_lng=P(),*a_lnb=P(),*a_qkvw=P(),*a_qkvbi=P(),*a_projw=P(),*a_projbi=P();
  // hybrid
  const float *h_dww=P(),*h_dwb=P(),*h_lng=P(),*h_lnb=P(),*h_pww=P(),*h_pwb=P(),
              *h_g1w=P(),*h_g1b=P(),*h_g2w=P(),*h_g2b=P(),*h_fw=P(),*h_fb=P();
  // freq (low, mid, high, spat : w,b,g,bb), a1, a2, fuse
  const float *f_cw[4],*f_cb[4],*f_cg[4],*f_cbb[4];
  for (int i=0;i<4;i++){ f_cw[i]=P(); f_cb[i]=P(); f_cg[i]=P(); f_cbb[i]=P(); }
  const float *f_a1w=P(),*f_a1b=P(),*f_a2w=P(),*f_a2b=P();
  const float *f_fw=P(),*f_fb=P(),*f_fg=P(),*f_fbb=P();
  // edge (sob, lap, grad, spat), a1, a2, fuse
  const float *e_cw[4],*e_cb[4],*e_cg[4],*e_cbb[4];
  for (int i=0;i<4;i++){ e_cw[i]=P(); e_cb[i]=P(); e_cg[i]=P(); e_cbb[i]=P(); }
  const float *e_a1w=P(),*e_a1b=P(),*e_a2w=P(),*e_a2b=P();
  const float *e_fw=P(),*e_fb=P(),*e_fg=P(),*e_fbb=P();
  // contrast
  const float *c_mw=P(),*c_mb=P(),*c_mg=P(),*c_mbb=P();
  const float *c_sw=P(),*c_sb=P(),*c_sg=P(),*c_sbb=P();
  const float *c_ldw=P(),*c_ldb=P(),*c_ldg=P(),*c_ldbb=P();
  const float *c_lpw=P(),*c_lpb=P(),*c_lpg=P(),*c_lpbb=P();
  const float *c_dw=P(),*c_db=P(),*c_dg=P(),*c_dbb=P();
  const float *c_b1w=P(),*c_b1b=P(),*c_b2w=P(),*c_b2b=P();
  const float *c_v1w=P(),*c_v1b=P(),*c_v1g=P(),*c_v1bb=P();
  const float *c_v2w=P(),*c_v2b=P(),*c_v2g=P(),*c_v2bb=P();
  const float *c_fw=P(),*c_fb=P(),*c_fg=P(),*c_fbb=P();
  // sem
  const float *s_cw[4],*s_cb[4],*s_cg[4],*s_cbb[4];
  for (int i=0;i<4;i++){ s_cw[i]=P(); s_cb[i]=P(); s_cg[i]=P(); s_cbb[i]=P(); }
  const float *s_fw=P(),*s_fb=P(),*s_fg=P(),*s_fbb=P();
  const float *s_a1w=P(),*s_a1b=P(),*s_a2w=P(),*s_a2b=P();
  // tex
  const float *t_dww=P(),*t_dwb=P(),*t_dwg=P(),*t_dwbb=P(),*t_pww=P(),*t_pwb=P();

  // ---- workspace carve-out ----
  char* ws = (char*)d_ws; size_t cur = 0;
  auto alloc = [&](size_t bytes) -> void* {
    void* p = ws + cur; cur = (cur + bytes + 255) & ~(size_t)255; return p;
  };
  const size_t nCHW = (size_t)BATCH * CHW;
  float* dense = (float*)alloc(BATCH * NEXP * sizeof(float));
  unsigned short* xb = (unsigned short*)alloc(nCHW * 2);
  float* eout = (float*)alloc((size_t)NEXP * nCHW * sizeof(float));
  float* sf = (float*)alloc(4 * nCHW * sizeof(float));     // f32 scratch slots
  float *t0 = sf, *t1 = sf + nCHW, *t2 = sf + 2*nCHW, *t3 = sf + 3*nCHW;
  unsigned short* sb = (unsigned short*)alloc(4 * nCHW * 2); // bf16 scratch slots
  unsigned short *s0b = sb, *s1b = sb + nCHW, *s2b = sb + 2*nCHW, *s3b = sb + 3*nCHW;
  float* gapv  = (float*)alloc(BATCH * DIMC * sizeof(float));
  float* gapv2 = (float*)alloc(BATCH * DIMC * sizeof(float));
  float* gate  = (float*)alloc(BATCH * DIMC * sizeof(float));
  float* pool  = (float*)alloc(BATCH * DIMC * 64 * sizeof(float));
  unsigned short* poolb = (unsigned short*)alloc(BATCH * DIMC * 64 * 2);
  float* pc    = (float*)alloc(BATCH * 32 * 64 * sizeof(float));
  unsigned short* wpool = (unsigned short*)alloc(640 * 1024); // bf16 weight pool
  unsigned short* wcur = wpool;

  const int EW = 256;
  int ewg = (int)((nCHW + EW - 1) / EW);
  auto packW = [&](const float* w, int n) -> unsigned short* {
    unsigned short* dst = wcur; wcur += n;
    k_pack<<<div_up(n, EW), EW, 0, stream>>>(w, dst, (size_t)n);
    return dst;
  };
  auto gemm = [&](const unsigned short* Wm, const unsigned short* X, const float* bias,
                  float* Y, unsigned short* Yb, int K, int Cout, int N,
                  int coff, int yctot, int act) {
    dim3 g(N / 32, Cout / 16, BATCH);
    k_gemm<<<g, 32, 0, stream>>>(Wm, X, bias, Y, Yb, K, Cout, N, K * N, coff, yctot, act);
  };
  auto ln = [&](const float* X, const float* g, const float* bb, float* Y,
                unsigned short* Yb, int C, int N, int act, int ycoff, int yctot) {
    int tot = BATCH * N;
    k_ln2d<<<div_up(tot, EW), EW, 0, stream>>>(X, g, bb, Y, Yb, C, N, act, ycoff, yctot);
  };

  // pack weights for all WMMA GEMMs
  unsigned short* wb_qkv = packW(a_qkvw, 3*DIMC*DIMC);
  unsigned short* wb_prj = packW(a_projw, DIMC*DIMC);
  unsigned short* wb_hpw = packW(h_pww, DIMC*DIMC);
  unsigned short* wb_hf  = packW(h_fw,  DIMC*DIMC);
  unsigned short *wb_fc[4], *wb_ec[4];
  for (int i=0;i<4;i++) wb_fc[i] = packW(f_cw[i], 32*DIMC);
  unsigned short* wb_ff = packW(f_fw, DIMC*DIMC);
  for (int i=0;i<4;i++) wb_ec[i] = packW(e_cw[i], 32*DIMC);
  unsigned short* wb_ef = packW(e_fw, DIMC*DIMC);
  unsigned short* wb_clp = packW(c_lpw, DIMC*DIMC);
  unsigned short* wb_cd  = packW(c_dw,  DIMC*2*DIMC);
  unsigned short* wb_cf  = packW(c_fw,  DIMC*DIMC);
  unsigned short* wb_sc0 = packW(s_cw[0], 32*DIMC);
  unsigned short* wb_sf = packW(s_fw, DIMC*DIMC);
  unsigned short* wb_tp = packW(t_pww, DIMC*DIMC);

  // x -> bf16; router
  k_pack<<<ewg, EW, 0, stream>>>(x, xb, nCHW);
  k_gap<<<div_up(BATCH*DIMC, EW), EW, 0, stream>>>(x, gapv, DIMC, HWN);
  k_router<<<1, 32, 0, stream>>>(gapv, r_w1, r_b1, r_w2, r_b2, dense);

  float* eo[NEXP];
  for (int e = 0; e < NEXP; e++) eo[e] = eout + (size_t)e * nCHW;

  // ---- expert 0: attention ----
  ln(x, a_lng, a_lnb, t3, s3b, DIMC, HWN, 0, 0, DIMC);
  gemm(wb_qkv, s3b, a_qkvbi, t0, s0b, DIMC, 3*DIMC, HWN, 0, 3*DIMC, 0); // qkv in t0..t2/s0b..s2b
  k_attn<<<dim3(HEADS, BATCH), 32, 0, stream>>>(s0b, t3, s3b);
  gemm(wb_prj, s3b, a_projbi, eo[0], nullptr, DIMC, DIMC, HWN, 0, DIMC, 0);
  k_add<<<ewg, EW, 0, stream>>>(eo[0], x, eo[0], nCHW);

  // ---- expert 1: hybrid ----
  k_dwconv<<<ewg, EW, 0, stream>>>(x, h_dww, h_dwb, t3, 5, 2);
  ln(t3, h_lng, h_lnb, t3, s3b, DIMC, HWN, 0, 0, DIMC);
  gemm(wb_hpw, s3b, h_pwb, t3, nullptr, DIMC, DIMC, HWN, 0, DIMC, 1); // gelu
  k_se<<<BATCH, 128, 0, stream>>>(gapv, h_g1w, h_g1b, h_g2w, h_g2b, gate, DIMC, 32);
  k_scalech<<<ewg, EW, 0, stream>>>(t3, gate, t3, s3b, DIMC, HWN);
  gemm(wb_hf, s3b, h_fb, eo[1], nullptr, DIMC, DIMC, HWN, 0, DIMC, 0);
  k_add<<<ewg, EW, 0, stream>>>(eo[1], x, eo[1], nCHW);

  // ---- expert 2: freq ----
  k_freqpre<<<ewg, EW, 0, stream>>>(x, s0b, s1b, s2b); // low, mid, high (bf16)
  {
    const unsigned short* src[4] = { s0b, s1b, s2b, xb };
    for (int i = 0; i < 4; i++) {
      gemm(wb_fc[i], src[i], f_cb[i], t0, nullptr, DIMC, 32, HWN, 0, 32, 0);
      ln(t0, f_cg[i], f_cbb[i], t3, s3b, 32, HWN, 1, i*32, DIMC); // cbr -> feats slice
    }
  }
  k_gap<<<div_up(BATCH*DIMC, EW), EW, 0, stream>>>(t3, gapv2, DIMC, HWN);
  k_se<<<BATCH, 128, 0, stream>>>(gapv2, f_a1w, f_a1b, f_a2w, f_a2b, gate, DIMC, 16);
  k_scalech<<<ewg, EW, 0, stream>>>(t3, gate, t3, s3b, DIMC, HWN);
  gemm(wb_ff, s3b, f_fb, t0, nullptr, DIMC, DIMC, HWN, 0, DIMC, 0);
  ln(t0, f_fg, f_fbb, eo[2], nullptr, DIMC, HWN, 1, 0, DIMC);
  k_add<<<ewg, EW, 0, stream>>>(eo[2], x, eo[2], nCHW);

  // ---- expert 3: edge ----
  k_edgefeat<<<ewg, EW, 0, stream>>>(x, s0b, s1b, s2b); // sobel, lap, grad (bf16)
  {
    const unsigned short* src[4] = { s0b, s1b, s2b, xb };
    for (int i = 0; i < 4; i++) {
      gemm(wb_ec[i], src[i], e_cb[i], t0, nullptr, DIMC, 32, HWN, 0, 32, 0);
      ln(t0, e_cg[i], e_cbb[i], t3, s3b, 32, HWN, 1, i*32, DIMC);
    }
  }
  k_gap<<<div_up(BATCH*DIMC, EW), EW, 0, stream>>>(t3, gapv2, DIMC, HWN);
  k_se<<<BATCH, 128, 0, stream>>>(gapv2, e_a1w, e_a1b, e_a2w, e_a2b, gate, DIMC, 16);
  k_scalech<<<ewg, EW, 0, stream>>>(t3, gate, t3, s3b, DIMC, HWN);
  gemm(wb_ef, s3b, e_fb, t0, nullptr, DIMC, DIMC, HWN, 0, DIMC, 0);
  ln(t0, e_fg, e_fbb, eo[3], nullptr, DIMC, HWN, 1, 0, DIMC);
  k_add<<<ewg, EW, 0, stream>>>(eo[3], x, eo[3], nCHW);

  // ---- expert 4: contrast ----
  k_dwconv<<<ewg, EW, 0, stream>>>(x, c_mw, c_mb, t0, 5, 2);
  ln(t0, c_mg, c_mbb, t0, nullptr, DIMC, HWN, 0, 0, DIMC);      // mean
  k_absdiff<<<ewg, EW, 0, stream>>>(x, t0, t1, nCHW);            // diff
  k_dwconv<<<ewg, EW, 0, stream>>>(t1, c_sw, c_sb, t2, 5, 2);
  ln(t2, c_sg, c_sbb, t2, nullptr, DIMC, HWN, 0, 0, DIMC);      // std
  k_dwconv<<<ewg, EW, 0, stream>>>(x, c_ldw, c_ldb, t3, 3, 1);
  ln(t3, c_ldg, c_ldbb, t3, s3b, DIMC, HWN, 1, 0, DIMC);        // cbr(lc_dw)
  gemm(wb_clp, s3b, c_lpb, t3, nullptr, DIMC, DIMC, HWN, 0, DIMC, 0);
  ln(t3, c_lpg, c_lpbb, t3, s0b, DIMC, HWN, 0, 0, 2*DIMC);      // lc -> cat[:,0:128] bf16
  k_pack_slice<<<ewg, EW, 0, stream>>>(t1, s0b, DIMC, HWN, DIMC, 2*DIMC); // diff -> cat[:,128:256]
  gemm(wb_cd, s0b, c_db, t3, nullptr, 2*DIMC, DIMC, HWN, 0, DIMC, 0);
  ln(t3, c_dg, c_dbb, t3, nullptr, DIMC, HWN, 1, 0, DIMC);      // cf = cbr(cat)
  k_gap<<<div_up(BATCH*DIMC, EW), EW, 0, stream>>>(t3, gapv2, DIMC, HWN);
  k_se<<<BATCH, 128, 0, stream>>>(gapv2, c_b1w, c_b1b, c_b2w, c_b2b, gate, DIMC, 32);
  k_scalech<<<ewg, EW, 0, stream>>>(t3, gate, t3, nullptr, DIMC, HWN);
  k_conv3x3<<<ewg, EW, 0, stream>>>(t3, c_v1w, c_v1b, t0, DIMC, DIMC);
  ln(t0, c_v1g, c_v1bb, t0, nullptr, DIMC, HWN, 1, 0, DIMC);    // vis1 = cbr 3x3
  k_conv3x3<<<ewg, EW, 0, stream>>>(t0, c_v2w, c_v2b, t1, DIMC, DIMC);
  ln(t1, c_v2g, c_v2bb, t1, nullptr, DIMC, HWN, 0, 0, DIMC);    // vis2 = conv+LN
  k_diveps<<<ewg, EW, 0, stream>>>(t1, t2, t1, nCHW);            // vis / (std+eps)
  k_pack<<<ewg, EW, 0, stream>>>(t1, s0b, nCHW);
  gemm(wb_cf, s0b, c_fb, t0, nullptr, DIMC, DIMC, HWN, 0, DIMC, 0);
  ln(t0, c_fg, c_fbb, eo[4], nullptr, DIMC, HWN, 1, 0, DIMC);
  k_add<<<ewg, EW, 0, stream>>>(eo[4], x, eo[4], nCHW);

  // ---- expert 5: semantic ----
  {
    const int kss[4] = {8, 4, 2, 1};
    for (int i = 0; i < 4; i++) {
      int kk = kss[i], n2 = kk * kk;
      int np = BATCH * DIMC * n2;
      k_apool<<<div_up(np, EW), EW, 0, stream>>>(x, pool, kk);
      if (kk == 8) {                                   // N=64 -> WMMA fast path
        k_pack<<<div_up(np, EW), EW, 0, stream>>>(pool, poolb, (size_t)np);
        gemm(wb_sc0, poolb, s_cb[0], pc, nullptr, DIMC, 32, 64, 0, 32, 0);
      } else {                                         // tiny N -> direct f32
        k_gemm_small<<<div_up(BATCH*32*n2, 64), 64, 0, stream>>>(
            s_cw[i], pool, s_cb[i], pc, DIMC, 32, n2, 0);
      }
      ln(pc, s_cg[i], s_cbb[i], pc, nullptr, 32, n2, 1, 0, 32);
      k_resize<<<div_up(BATCH*32*HWN, EW), EW, 0, stream>>>(pc, kk, 32, t3, s3b, i*32, DIMC);
    }
  }
  gemm(wb_sf, s3b, s_fb, t0, nullptr, DIMC, DIMC, HWN, 0, DIMC, 0);
  ln(t0, s_fg, s_fbb, t0, nullptr, DIMC, HWN, 1, 0, DIMC);      // f = cbr(cat)
  k_gap<<<div_up(BATCH*DIMC, EW), EW, 0, stream>>>(t0, gapv2, DIMC, HWN);
  k_se<<<BATCH, 128, 0, stream>>>(gapv2, s_a1w, s_a1b, s_a2w, s_a2b, gate, DIMC, 16);
  k_scalech<<<ewg, EW, 0, stream>>>(t0, gate, eo[5], nullptr, DIMC, HWN);
  k_add<<<ewg, EW, 0, stream>>>(eo[5], x, eo[5], nCHW);

  // ---- expert 6: texture ----
  k_dwconv<<<ewg, EW, 0, stream>>>(x, t_dww, t_dwb, t0, 3, 1);
  ln(t0, t_dwg, t_dwbb, t0, s0b, DIMC, HWN, 1, 0, DIMC);
  gemm(wb_tp, s0b, t_pwb, eo[6], nullptr, DIMC, DIMC, HWN, 0, DIMC, 0);
  k_add<<<ewg, EW, 0, stream>>>(eo[6], x, eo[6], nCHW);

  // ---- weighted combine ----
  k_combine<<<ewg, EW, 0, stream>>>(eout, dense, (float*)d_out);
}
